// ThrusterLag_71399536329340
// MI455X (gfx1250) — matrix operand order repaired and checked by reading, not compiled, not run on hardware
//
#include <hip/hip_runtime.h>
#include <hip/hip_fp16.h>

// ---------------------------------------------------------------------------
// ThrusterLag: y[t] = a*y[t-1] + (1-a)*tanh(2*dz((u-7.5)/2.5)),  a = exp(-dt/tau)
// Strategy (MI455X / gfx1250):
//   * 1 workgroup (8 waves, 256 thr) per batch; whole 8192x8 f32 series (256KB)
//     lives in LDS (<=320KB/WGP).  Read HBM once, write HBM once (~11us floor
//     at 23.3 TB/s for 256 MiB of traffic).
//   * scan-as-matmul: 16-step chunk = lower-triangular-ones 16x16 matmul via
//     v_wmma_f32_16x16x32_f16 with exact diagonal pre/post scaling (makes the
//     matrix channel-independent).
//   * carries resolved on-chip: register carry chunk->chunk (wave32 __shfl),
//     LDS carry table across the 16 x 512-step sub-segments, fused fix-up+store.
// ---------------------------------------------------------------------------

typedef __attribute__((ext_vector_type(16))) _Float16 v16h;
typedef __attribute__((ext_vector_type(8)))  float    v8f;
typedef int vint4 __attribute__((vector_size(16)));    // type expected by async builtin

#define L_STEPS 8192
#define N_CH    8
#define SEG     512                      // sub-segment length (16 per batch)
#define NSUB    (L_STEPS / SEG)          // 16
#define DWPB    (L_STEPS * N_CH)         // dwords per batch = 65536
#define SMEM_FLOATS (DWPB + NSUB*N_CH + NSUB*N_CH + N_CH)   // data + e + C + lam

__device__ __forceinline__ float tanh_f(float x) {
#if __has_builtin(__builtin_amdgcn_tanhf)
  return __builtin_amdgcn_tanhf(x);      // native v_tanh_f32 (TRANS32 on CDNA5)
#else
  return tanhf(x);
#endif
}

__device__ __forceinline__ float nlmap(float uraw) {
  float uu = fmaf(uraw, 0.4f, -3.0f);            // (u - 7.5) / 2.5
  float ad = fmaxf(fabsf(uu) - 0.05f, 0.0f);     // deadzone
  float ud = copysignf(ad, uu);
  return tanh_f(2.0f * ud);                      // saturation
}

__global__ void thruster_lag_scan(const float* __restrict__ u,
                                  const float* __restrict__ tau_param,
                                  float* __restrict__ out) {
  extern __shared__ float smem[];                // [0,65536): x then y (in place)
  float* eTab = smem + DWPB;                     // NSUB*8 sub-segment end values
  float* cTab = eTab + NSUB * N_CH;              // NSUB*8 carry-in values
  float* lamS = cTab + NSUB * N_CH;              // 8 per-channel ln(a)

  const int tid  = threadIdx.x;
  const int b    = blockIdx.x;
  const int lane = tid & 31;
  const int wv   = tid >> 5;                     // wave id 0..7
  const bool lo  = lane < 16;
  const int  n   = lane & 15;                    // WMMA column 0..15
  const int  c   = n & 7;                        // channel
  const int  scol = 2 * wv + (n >> 3);           // this column's sub-segment

  const float* gin = u + (size_t)b * DWPB;

  // ---- stage input batch into LDS (async copy path on gfx1250) ----
#if __has_builtin(__builtin_amdgcn_global_load_async_to_lds_b128) && \
    __has_builtin(__builtin_amdgcn_s_wait_asynccnt)
  {
#pragma unroll 4
    for (int j = 0; j < DWPB / 4 / 256; ++j) {
      int q = tid + 256 * j;
      __builtin_amdgcn_global_load_async_to_lds_b128(
          (__attribute__((address_space(1))) vint4*)(gin + 4 * q),
          (__attribute__((address_space(3))) vint4*)(smem + 4 * q),
          0, 0);
    }
    __builtin_amdgcn_s_wait_asynccnt(0);
  }
#else
#pragma unroll 4
  for (int j = 0; j < DWPB / 4 / 256; ++j) {
    int q = tid + 256 * j;
    ((float4*)smem)[q] = ((const float4*)gin)[q];
  }
#endif
  __syncthreads();

  // ---- per-lane filter constants ----
  float p    = tau_param[c];
  float tau  = log1pf(__expf(p)) + 0.01f;        // softplus + TAU_MIN
  float lam  = -0.01f / tau;                     // ln(a)  (a = exp(-DT/tau))
  float a1   = __expf(lam);
  float ainv = __expf(-lam);
  float oma  = 1.0f - a1;
  float a9   = __expf(9.0f * lam);               // a^9
  float w0   = oma * __expf(8.0f * lam);         // (1-a)*a^8 : prescale seed
  float powA[8];
#pragma unroll
  for (int r = 0; r < 8; ++r)                    // a^(row-8): rows lo=0..7 hi=8..15
    powA[r] = __expf((float)(r + (lo ? -8 : 0)) * lam);

  // ---- constant lower-triangular ones matrix A (16x32 f16, K=16..31 zero) ----
  v16h Av;
#pragma unroll
  for (int h = 0; h < 16; ++h) Av[h] = (_Float16)0.0f;
  {
    int i  = n;                                  // my row
    int jb = lo ? 0 : 32;                        // lo halves 0..15 cover K=0..15; hi zero
#pragma unroll
    for (int h = 0; h < 16; ++h)
      Av[h] = (jb + h <= i) ? (_Float16)1.0f : (_Float16)0.0f;
  }

  // ---- phase A: local scan of my column's 512-step sub-segment ----
  float carry = 0.0f;
  if (scol == 0) carry = nlmap(smem[c]);         // reference: y[0] = x_nl[0]

  for (int kk = 0; kk < SEG / 16; ++kk) {
    const int tbase = scol * SEG + kk * 16;

    // B operand: prescaled x' = (1-a)*a^(8-j)*x_nl  (K rows = time-in)
    v16h Bv;
#pragma unroll
    for (int h = 0; h < 16; ++h) Bv[h] = (_Float16)0.0f;
    if (lo) {
      float wj = w0;
#pragma unroll
      for (int r = 0; r < 8; ++r) {
        float x0 = smem[(tbase + 2 * r    ) * N_CH + c];
        float x1 = smem[(tbase + 2 * r + 1) * N_CH + c];
        Bv[2 * r    ] = (_Float16)(nlmap(x0) * wj); wj *= ainv;
        Bv[2 * r + 1] = (_Float16)(nlmap(x1) * wj); wj *= ainv;
      }
    }

    v8f D;
#pragma unroll
    for (int r = 0; r < 8; ++r) D[r] = 0.0f;
    D = __builtin_amdgcn_wmma_f32_16x16x32_f16(false, Av, false, Bv,
                                               (short)0, D, false, false);

    // postscale + in-place store; y[j] = a^(j-8)*(D[j] + a^9*carry)
    float c9 = a9 * carry;
    float y7 = 0.0f;
#pragma unroll
    for (int r = 0; r < 8; ++r) {
      int irow = r + (lo ? 0 : 8);
      float y  = powA[r] * (D[r] + c9);
      smem[(tbase + irow) * N_CH + c] = y;
      if (r == 7) y7 = y;
    }
    // next chunk's carry = row 15 of this column (lives in hi-half lane)
    carry = __shfl(y7, n + 16, 32);
  }
  if (lo) eTab[scol * N_CH + c] = carry;         // sub-segment end values
  __syncthreads();

  // ---- phase B-pre: sub-segment carry table  C_s = sum a^{512(s-1-v)} e_v ----
  if (tid < NSUB * N_CH) {
    int s = tid >> 3, cc = tid & 7;
    float pp = tau_param[cc];
    float tt = log1pf(__expf(pp)) + 0.01f;
    float lm = -0.01f / tt;
    float alpha = __expf((float)SEG * lm);       // a^512
    float acc = 0.0f;
    for (int v = 0; v < s; ++v) acc = acc * alpha + eTab[v * N_CH + cc];
    cTab[tid] = acc;
    if (tid < N_CH) lamS[tid] = lm;
  }
  __syncthreads();

  // ---- phase B: fix-up y += a^(toff+1)*C_s fused with coalesced b128 store ----
  float4* og = (float4*)(out + (size_t)b * DWPB);
  for (int j = 0; j < DWPB / 4 / 256; ++j) {
    int q = tid + 256 * j;
    float4 y4 = ((const float4*)smem)[q];
    int t = q >> 1;                              // 2 float4 per time step (8 ch)
    int c0 = (q & 1) * 4;
    int s = t >> 9;
    float pf = (float)((t & (SEG - 1)) + 1);     // toff + 1
    float* yp = &y4.x;
#pragma unroll
    for (int i = 0; i < 4; ++i)
      yp[i] += __expf(pf * lamS[c0 + i]) * cTab[s * N_CH + c0 + i];
    og[q] = y4;
  }
}

extern "C" void kernel_launch(void* const* d_in, const int* in_sizes, int n_in,
                              void* d_out, int out_size, void* d_ws, size_t ws_size,
                              hipStream_t stream) {
  const float* u         = (const float*)d_in[0];
  const float* tau_param = (const float*)d_in[1];
  float* out = (float*)d_out;

  int nbatch = in_sizes[0] / DWPB;               // 512
  size_t smem_bytes = (size_t)SMEM_FLOATS * sizeof(float);   // ~257 KiB < 320 KiB

  (void)hipFuncSetAttribute(reinterpret_cast<const void*>(thruster_lag_scan),
                            hipFuncAttributeMaxDynamicSharedMemorySize,
                            (int)smem_bytes);
  thruster_lag_scan<<<nbatch, 256, smem_bytes, stream>>>(u, tau_param, out);
}
